// CrossConv2d_48876727828962
// MI455X (gfx1250) — compile-verified, matrix operand order checked by reading, not run on hardware
//
#include <hip/hip_runtime.h>

// ---- CDNA5 (gfx1250) types for WMMA -------------------------------------
typedef __attribute__((ext_vector_type(16))) __bf16 v16bf;
typedef __attribute__((ext_vector_type(8)))  float  v8f;

#define Hh   128
#define Ww   128
#define C1c  64
#define C2c  64
#define CCc  128
#define Ss   8
#define BSs  32            // B*S
#define TILE_N 64          // output pixels per workgroup (x direction)
#define NPX  (TILE_N + 2)  // staged pixels incl. 1-px halo each side (66)
#define NT   (TILE_N / 16) // n-tiles per wave (4)
#define WPACK_ELEMS (9 * 4 * CCc * 2 * 16)   // 147456 bf16 = 288 KB
#define EPSf 1e-5f

__device__ __forceinline__ unsigned short f2bf(float f) {
  unsigned int x = __builtin_bit_cast(unsigned int, f);
  x += 0x7FFFu + ((x >> 16) & 1u);          // round-to-nearest-even
  return (unsigned short)(x >> 16);
}

// -------------------------------------------------------------------------
// One-shot pack: w (OIHW fp32) -> bf16 in WMMA A-fragment order:
//   flat = (((kk*4 + icb)*128 + oc)*2 + h)*16 + t
// so that lane L of a wave reads its whole 32-byte A operand contiguously:
//   lanes 0-15 (h=0): K = t0..7 and 16+t0..23 ; lanes 16-31 (h=1): K = 8.. / 24..
// -------------------------------------------------------------------------
__global__ __launch_bounds__(256)
void pack_weights(const float* __restrict__ w, unsigned short* __restrict__ wp) {
  const int idx = blockIdx.x * 256 + threadIdx.x;
  if (idx >= WPACK_ELEMS) return;
  const int t   = idx & 15;
  const int h   = (idx >> 4) & 1;
  const int oc  = (idx >> 5) & (CCc - 1);
  const int icb = (idx >> 12) & 3;
  const int kk  = idx >> 14;
  const int t0 = t & 7, b4 = t >> 3;
  const int k  = 16 * b4 + 8 * h + t0;        // K within the 32-wide block
  const int ic = icb * 32 + k;
  const int ky = kk / 3, kx = kk % 3;
  wp[idx] = f2bf(w[(((size_t)oc * CCc + ic) * 3 + ky) * 3 + kx]);
}

// -------------------------------------------------------------------------
// Main kernel: implicit-GEMM 3x3 conv + BN + skip + ReLU.
// Block = 256 threads (8 wave32): 128 oc x 64 px of one (bs, y) row.
// K = 128 ic x 9 taps, consumed as 9 (ky,kx) slices x 4 ic-blocks of 32.
// A operands come straight from packed global (bf16, L2-resident).
// B operands come from LDS staged in fragment-ready order (32B reads).
// -------------------------------------------------------------------------
__global__ __launch_bounds__(256)
void cross_conv_wmma(const float* __restrict__ u,
                     const float* __restrict__ v,
                     const unsigned short* __restrict__ wp,
                     const float* __restrict__ gamma,
                     const float* __restrict__ beta,
                     const float* __restrict__ mean,
                     const float* __restrict__ var,
                     float* __restrict__ out)
{
  // [row r][ic-block][pixel j][lane-half h][16 bf16]  = 50,688 B
  __shared__ __align__(32) unsigned short sInF[3][4][NPX][2][16];

  const int tid  = threadIdx.x;
  const int lane = tid & 31;
  const int wave = tid >> 5;

  const int x0 = blockIdx.x * TILE_N;   // 0 or 64
  const int y  = blockIdx.y;            // 0..127
  const int bs = blockIdx.z;            // 0..31
  const int b  = bs / Ss;
  const int s  = bs % Ss;

  // ---- stage 3 input rows (y-1, y, y+1) as bf16 fragments, zero-padded --
  for (int idx = tid; idx < 3 * CCc * NPX; idx += 256) {
    const int j = idx % NPX;                   // coalesced along x
    const int c = (idx / NPX) & (CCc - 1);
    const int r = idx / (NPX * CCc);
    const int yy = y + r - 1;
    const int xx = x0 - 1 + j;
    float val = 0.0f;
    if ((unsigned)yy < (unsigned)Hh && (unsigned)xx < (unsigned)Ww) {
      if (c < C1c)
        val = u[(((size_t)b * C1c + c) * Hh + yy) * Ww + xx];
      else
        val = v[((((size_t)b * Ss + s) * C2c + (c - C1c)) * Hh + yy) * Ww + xx];
    }
    const int icb  = c >> 5;
    const int k    = c & 31;
    const int h    = (k >> 3) & 1;
    const int slot = (k & 7) | (((k >> 4) & 1) << 3);
    sInF[r][icb][j][h][slot] = f2bf(val);
  }
  __syncthreads();   // sInF is read-only below: only one barrier needed

  v8f acc[NT] = {};
  const int m_base = wave * 16;         // 16 output channels per wave
  const int am = m_base + (lane & 15);  // A row (oc) for this lane
  const int hH = lane >> 4;             // lane half selects K subset
  const int bn = lane & 15;             // B column (pixel) for this lane

  for (int kk = 0; kk < 9; ++kk) {
    const int ky = kk / 3, kx = kk % 3;
    if (kk < 8)   // pull next weight slice toward L1 (global_prefetch_b8)
      __builtin_prefetch(wp + (((size_t)(kk + 1) * 4) * CCc * 2 + tid) * 16, 0, 1);
    #pragma unroll
    for (int icb = 0; icb < 4; ++icb) {
      // A fragment: one aligned 32-byte vector load from packed weights
      const v16bf a = *(const v16bf*)(
          wp + ((((size_t)(kk * 4 + icb) * CCc + am) * 2 + hH) << 4));
      #pragma unroll
      for (int nt = 0; nt < NT; ++nt) {
        // B fragment: one aligned 32-byte LDS read (ds_load_b128 x2)
        const v16bf bfrag =
            *(const v16bf*)&sInF[ky][icb][nt * 16 + bn + kx][hH][0];
        acc[nt] = __builtin_amdgcn_wmma_f32_16x16x32_bf16(
            false, a, false, bfrag, (short)0, acc[nt], false, false);
      }
    }
  }

  // ---- epilogue: BN + skip + ReLU, fp32 store ---------------------------
  // C/D layout: VGPR r -> M = r (lanes 0-15) or r+8 (lanes 16-31); N = lane&15
  const int n_lane = lane & 15;
  const int m_off  = (lane >> 4) * 8;

  #pragma unroll
  for (int r = 0; r < 8; ++r) {
    const int oc = m_base + m_off + r;
    const float sc = gamma[oc] * rsqrtf(var[oc] + EPSf);
    const float sh = beta[oc] - mean[oc] * sc;

    #pragma unroll
    for (int nt = 0; nt < NT; ++nt) {
      const int x = x0 + nt * 16 + n_lane;
      float skip;
      if (oc < C1c)
        skip = u[(((size_t)b * C1c + oc) * Hh + y) * Ww + x];
      else
        skip = v[((((size_t)b * Ss + s) * C2c + (oc - C1c)) * Hh + y) * Ww + x];
      float val = fmaf(acc[nt][r], sc, sh) + skip;
      val = val > 0.0f ? val : 0.0f;
      out[(((size_t)bs * CCc + oc) * Hh + y) * Ww + x] = val;
    }
  }
}

extern "C" void kernel_launch(void* const* d_in, const int* in_sizes, int n_in,
                              void* d_out, int out_size, void* d_ws, size_t ws_size,
                              hipStream_t stream) {
  (void)in_sizes; (void)n_in; (void)out_size; (void)ws_size;
  const float* u     = (const float*)d_in[0];
  const float* v     = (const float*)d_in[1];
  const float* cw    = (const float*)d_in[2];
  const float* gamma = (const float*)d_in[3];
  const float* beta  = (const float*)d_in[4];
  const float* mean  = (const float*)d_in[5];
  const float* var   = (const float*)d_in[6];
  float* out = (float*)d_out;
  unsigned short* wp = (unsigned short*)d_ws;   // 288 KB packed bf16 weights

  pack_weights<<<dim3((WPACK_ELEMS + 255) / 256), dim3(256), 0, stream>>>(cw, wp);

  dim3 grid(Ww / TILE_N, Hh, BSs);   // (2, 128, 32) = 8192 workgroups
  dim3 block(256);                   // 8 wave32 waves
  cross_conv_wmma<<<grid, block, 0, stream>>>(u, v, wp, gamma, beta, mean, var, out);
}
